// PSTConvTranspose_52913997087088
// MI455X (gfx1250) — matrix operand order repaired and checked by reading, not compiled, not run on hardware
//
#include <hip/hip_runtime.h>
#include <hip/hip_bf16.h>

// ---------------- static problem configuration ----------------
#define B_     8
#define L1_    7
#define L2_    3
#define N1_    2048
#define N2_    512
#define CIN_   512
#define MID_   256
#define OUT_   512
#define CORIG_ 128
#define KT_    3
#define OCH_   (KT_ * MID_)          // 768
#define NFCH_  (MID_ + CORIG_)       // 384

// seed map derived from middles = [2,4,6], RADIUS=1, PAD=(0,0)
__constant__ int c_seedCnt[L1_]    = {1, 1, 2, 1, 2, 1, 1};
__constant__ int c_seedT2[L1_][2]  = {{0,0},{0,0},{0,1},{1,0},{1,2},{2,0},{2,0}};
__constant__ int c_seedK [L1_][2]  = {{0,0},{1,0},{2,0},{1,0},{2,0},{1,0},{2,0}};

// ---------------- WMMA / TDM types ----------------
typedef __attribute__((ext_vector_type(16))) __bf16 v16bf;
typedef __attribute__((ext_vector_type(8)))  float  v8f;
typedef __attribute__((ext_vector_type(4)))  unsigned int u32x4;
typedef __attribute__((ext_vector_type(8)))  unsigned int u32x8;

union FragAB { v16bf v; unsigned int u[8]; };

__device__ __forceinline__ unsigned short f2bf(float f) {
  unsigned int u = __builtin_bit_cast(unsigned int, f);
  unsigned int r = u + 0x7FFFu + ((u >> 16) & 1u);   // round-to-nearest-even
  return (unsigned short)(r >> 16);
}

// ---------------- TDM 2-D tile load: global (row-major, bf16) -> LDS ----------
// Tile: k_len elements along dim0 (contiguous), rows along dim1 with stride
// row_stride elements. LDS destination padded by 4 DWORDs every 16 DWORDs
// (64B rows -> 80B pitch == 40 halfs), matching fragment addressing below.
__device__ __forceinline__ void tdm_load_2d(
    unsigned lds_off, const unsigned short* gptr,
    unsigned k_len, unsigned rows,
    unsigned tensor_d0, unsigned row_stride, unsigned tensor_d1) {
  const unsigned long long ga = (unsigned long long)(size_t)gptr;
  u32x4 g0;
  g0.x = 1u;                                  // count=1, user descriptor
  g0.y = lds_off;                             // lds_addr (bytes)
  g0.z = (unsigned)ga;                        // global_addr[31:0]
  g0.w = (unsigned)(ga >> 32) | (2u << 30);   // global_addr[56:32] | type=2
  u32x8 g1;
  g1.s0 = (1u << 16)        // data_size = 1 -> 2 bytes
        | (1u << 20)        // pad_enable
        | (3u << 22)        // pad_interval: 16 DWORDs (64B)
        | (3u << 25);       // pad_amount:   4 DWORDs (16B)
  g1.s1 = (tensor_d0 & 0xFFFFu) << 16;                       // tensor_dim0[15:0]
  g1.s2 = (tensor_d0 >> 16) | ((tensor_d1 & 0xFFFFu) << 16); // dim0 hi | dim1 lo
  g1.s3 = (tensor_d1 >> 16) | (k_len << 16);                 // dim1 hi | tile_dim0
  g1.s4 = rows & 0xFFFFu;                                    // tile_dim1, tile_dim2=0
  g1.s5 = row_stride;                                        // tensor_dim0_stride[31:0]
  g1.s6 = 0u;                                                // stride hi | dim1_stride lo
  g1.s7 = 0u;
  asm volatile("tensor_load_to_lds %0, %1" :: "s"(g0), "s"(g1) : "memory");
}

// ---------------- f32 -> bf16 convert ----------------
__global__ __launch_bounds__(256) void f32_to_bf16_kernel(
    const float* __restrict__ src, unsigned short* __restrict__ dst, size_t n) {
  size_t i = (size_t)blockIdx.x * 256 + threadIdx.x;
  if (i < n) dst[i] = f2bf(src[i]);
}

// ---------------- batched f32 -> bf16 transpose (R x C) -> (C x R) ----------
__global__ __launch_bounds__(256) void transpose_to_bf16_kernel(
    const float* __restrict__ src, unsigned short* __restrict__ dst, int R, int C) {
  const int gid = blockIdx.x * 256 + threadIdx.x;
  if (gid >= R * C) return;
  const size_t base = (size_t)blockIdx.z * R * C;
  const int r = gid / C;
  const int c = gid % C;
  dst[base + (size_t)c * R + r] = f2bf(src[base + gid]);
}

// ---------------- batched bf16 WMMA GEMM with TDM double-buffered staging ----
// C[M x N] = A[M x K] * Bt^T, A row-major M x K, Bt row-major N x K (pre-
// transposed), C row-major. M%64==0, N%128==0, K%32==0.
// Block: 256 threads (8 waves), block tile 64x128, wave tile 32x32 (2x2 WMMA).
__global__ __launch_bounds__(256) void wmma_gemm_bf16_kernel(
    const unsigned short* __restrict__ A, const unsigned short* __restrict__ Bt,
    float* __restrict__ C, int M, int N, int K,
    long long sAb, long long sBb, long long sCb) {
  __shared__ unsigned short As[2][64 * 40];    // 80B row pitch (TDM-padded)
  __shared__ unsigned short Bs[2][128 * 40];

  const unsigned short* Ab = A  + (size_t)blockIdx.z * sAb;
  const unsigned short* Bb = Bt + (size_t)blockIdx.z * sBb;
  float*                Cb = C  + (size_t)blockIdx.z * sCb;

  const int blockM = blockIdx.y * 64;
  const int blockN = blockIdx.x * 128;
  const int tid    = threadIdx.x;
  const int wave   = tid >> 5;
  const int lane   = tid & 31;
  const int laneM  = lane & 15;
  const int laneHi = lane >> 4;
  const int waveM  = wave >> 2;   // 0..1
  const int waveN  = wave & 3;    // 0..3

  const v8f vzero = {0.f,0.f,0.f,0.f,0.f,0.f,0.f,0.f};
  v8f acc[2][2];
  acc[0][0] = vzero; acc[0][1] = vzero; acc[1][0] = vzero; acc[1][1] = vzero;

  const unsigned ldsA[2] = {(unsigned)(size_t)(&As[0][0]), (unsigned)(size_t)(&As[1][0])};
  const unsigned ldsB[2] = {(unsigned)(size_t)(&Bs[0][0]), (unsigned)(size_t)(&Bs[1][0])};

  // prologue: prime buffer 0
  if (wave == 0) {
    tdm_load_2d(ldsA[0], Ab + (size_t)blockM * K, 32u, 64u,  (unsigned)K, (unsigned)K, (unsigned)M);
    tdm_load_2d(ldsB[0], Bb + (size_t)blockN * K, 32u, 128u, (unsigned)K, (unsigned)K, (unsigned)N);
  }

  int buf = 0;
  for (int k0 = 0; k0 < K; k0 += 32) {
    if (wave == 0) {
      if (k0 + 32 < K) {   // issue next slab into the other buffer, then wait
        const int nb = buf ^ 1;
        tdm_load_2d(ldsA[nb], Ab + (size_t)blockM * K + (k0 + 32), 32u, 64u,  (unsigned)K, (unsigned)K, (unsigned)M);
        tdm_load_2d(ldsB[nb], Bb + (size_t)blockN * K + (k0 + 32), 32u, 128u, (unsigned)K, (unsigned)K, (unsigned)N);
        __builtin_amdgcn_s_wait_tensorcnt(2);   // current buffer's pair done
      } else {
        __builtin_amdgcn_s_wait_tensorcnt(0);
      }
    }
    __syncthreads();

    const unsigned short* Asb = As[buf];
    const unsigned short* Bsb = Bs[buf];

    // fragments per documented CDNA5 layouts (K-pairs -> one b32 each)
    FragAB af[2], bfr[2];
    #pragma unroll
    for (int mi = 0; mi < 2; ++mi) {
      const int row = waveM * 32 + mi * 16 + laneM;
      #pragma unroll
      for (int j = 0; j < 8; ++j) {
        const int kk = ((j < 4) ? (2 * j) : (16 + 2 * (j - 4))) + laneHi * 8;
        af[mi].u[j] = *(const unsigned int*)(&Asb[row * 40 + kk]);
      }
    }
    #pragma unroll
    for (int ni = 0; ni < 2; ++ni) {
      const int col = waveN * 32 + ni * 16 + laneM;
      #pragma unroll
      for (int j = 0; j < 8; ++j) {
        const int kk = ((j < 4) ? (2 * j) : (16 + 2 * (j - 4))) + laneHi * 8;
        bfr[ni].u[j] = *(const unsigned int*)(&Bsb[col * 40 + kk]);
      }
    }

    #pragma unroll
    for (int mi = 0; mi < 2; ++mi)
      #pragma unroll
      for (int ni = 0; ni < 2; ++ni)
        acc[mi][ni] = __builtin_amdgcn_wmma_f32_16x16x32_bf16(
            false, af[mi].v, false, bfr[ni].v, (short)0, acc[mi][ni], false, false);

    __syncthreads();   // all waves done reading before this buffer is re-filled
    buf ^= 1;
  }

  // store: D VGPR r -> row r (lanes 0-15) / r+8 (lanes 16-31)
  #pragma unroll
  for (int mi = 0; mi < 2; ++mi)
    #pragma unroll
    for (int ni = 0; ni < 2; ++ni)
      #pragma unroll
      for (int r = 0; r < 8; ++r) {
        const int row = blockM + waveM * 32 + mi * 16 + r + laneHi * 8;
        const int col = blockN + waveN * 32 + ni * 16 + laneM;
        Cb[(size_t)row * N + col] = acc[mi][ni][r];
      }
}

// ---------------- BatchNorm batch statistics ----------------
__global__ __launch_bounds__(256) void bn_stats_kernel(
    const float* __restrict__ tf, float* __restrict__ meanp, float* __restrict__ rstdp) {
  __shared__ float s_sum[256];
  __shared__ float s_sq[256];
  const int cid = blockIdx.x;
  const int t2  = cid / OCH_;
  const int oc  = cid % OCH_;
  const int tid = threadIdx.x;
  float s = 0.f, ss = 0.f;
  for (int i = tid; i < B_ * N2_; i += 256) {
    const int b = i >> 9, n2 = i & (N2_ - 1);
    const float v = tf[(((size_t)b * L2_ + t2) * OCH_ + oc) * N2_ + n2];
    s += v; ss += v * v;
  }
  s_sum[tid] = s; s_sq[tid] = ss;
  __syncthreads();
  for (int str = 128; str > 0; str >>= 1) {
    if (tid < str) { s_sum[tid] += s_sum[tid + str]; s_sq[tid] += s_sq[tid + str]; }
    __syncthreads();
  }
  if (tid == 0) {
    const float inv = 1.f / (float)(B_ * N2_);
    const float m = s_sum[0] * inv;
    const float v = s_sq[0] * inv - m * m;
    meanp[cid] = m;
    rstdp[cid] = rsqrtf(v + 1e-5f);
  }
}

// ---------------- 3-NN search ----------------
__global__ __launch_bounds__(256) void three_nn_kernel(
    const float* __restrict__ oxyz, const float* __restrict__ xyzs,
    int* __restrict__ idxp, float* __restrict__ wgtp) {
  const int gid = blockIdx.x * 256 + threadIdx.x;
  if (gid >= B_ * L1_ * N1_) return;
  const int n  = gid % N1_;
  const int bt = gid / N1_;
  const int t1 = bt % L1_;
  const int b  = bt / L1_;

  const float* ap = oxyz + ((size_t)bt * N1_ + n) * 3;
  const float ax = ap[0], ay = ap[1], az = ap[2];

  float d[3]  = {3.4e38f, 3.4e38f, 3.4e38f};
  int   id[3] = {0, 0, 0};
  const int S = c_seedCnt[t1];
  for (int s = 0; s < S; ++s) {
    const int t2 = c_seedT2[t1][s];
    const float* sp = xyzs + ((size_t)b * L2_ + t2) * N2_ * 3;
    for (int j = 0; j < N2_; ++j) {
      const float dx = ax - sp[j * 3 + 0];
      const float dy = ay - sp[j * 3 + 1];
      const float dz = az - sp[j * 3 + 2];
      const float dd = dx * dx + dy * dy + dz * dz;
      const int m = s * N2_ + j;
      if (dd < d[0]) {
        d[2] = d[1]; id[2] = id[1];
        d[1] = d[0]; id[1] = id[0];
        d[0] = dd;   id[0] = m;
      } else if (dd < d[1]) {
        d[2] = d[1]; id[2] = id[1];
        d[1] = dd;   id[1] = m;
      } else if (dd < d[2]) {
        d[2] = dd;   id[2] = m;
      }
    }
  }
  float w[3], tot = 0.f;
  #pragma unroll
  for (int q = 0; q < 3; ++q) {
    const float di = sqrtf(fmaxf(d[q], 1e-12f));
    w[q] = 1.f / (di + 1e-8f);
    tot += w[q];
  }
  #pragma unroll
  for (int q = 0; q < 3; ++q) {
    wgtp[(size_t)gid * 3 + q] = w[q] / tot;
    idxp[(size_t)gid * 3 + q] = id[q];
  }
}

// ----- fused gather + BN/ReLU + interp + concat -> nfT (bf16, N1 x NFCH) -----
__global__ __launch_bounds__(256) void build_nf_kernel(
    const float* __restrict__ tf, const float* __restrict__ meanp,
    const float* __restrict__ rstdp, const float* __restrict__ gamma,
    const float* __restrict__ beta, const float* __restrict__ ofeat,
    const int* __restrict__ idxp, const float* __restrict__ wgtp,
    unsigned short* __restrict__ nf) {
  const size_t total = (size_t)B_ * L1_ * N1_ * NFCH_;
  const size_t gid = (size_t)blockIdx.x * 256 + threadIdx.x;
  if (gid >= total) return;
  const int c  = (int)(gid % NFCH_);
  const size_t r0 = gid / NFCH_;
  const int n  = (int)(r0 % N1_);
  const int bt = (int)(r0 / N1_);
  const int t1 = bt % L1_;
  const int b  = bt / L1_;

  float val;
  if (c < MID_) {
    const size_t pbase = ((size_t)bt * N1_ + n) * 3;
    const float g = gamma[c], be = beta[c];
    float acc = 0.f;
    #pragma unroll
    for (int q = 0; q < 3; ++q) {
      const int m  = idxp[pbase + q];
      const float w = wgtp[pbase + q];
      const int s  = m >> 9;            // N2 = 512
      const int n2 = m & (N2_ - 1);
      const int t2 = c_seedT2[t1][s];
      const int kk = c_seedK[t1][s];
      const int cid = (t2 * KT_ + kk) * MID_ + c;
      float v = tf[(((size_t)b * L2_ + t2) * OCH_ + (kk * MID_ + c)) * N2_ + n2];
      v = (v - meanp[cid]) * rstdp[cid] * g + be;
      v = fmaxf(v, 0.f);
      acc += w * v;
    }
    val = acc;
  } else {
    val = ofeat[((size_t)bt * CORIG_ + (c - MID_)) * N1_ + n];
  }
  nf[gid] = f2bf(val);   // gid == ((bt*N1 + n)*NFCH + c): transposed layout
}

// ---------------- launcher ----------------
extern "C" void kernel_launch(void* const* d_in, const int* in_sizes, int n_in,
                              void* d_out, int out_size, void* d_ws, size_t ws_size,
                              hipStream_t stream) {
  const float* xyzs  = (const float*)d_in[0];   // (B, L2, N2, 3)
  const float* oxyzs = (const float*)d_in[1];   // (B, L1, N1, 3)
  const float* feats = (const float*)d_in[2];   // (B, L2, CIN, N2)
  const float* ofeat = (const float*)d_in[3];   // (B, L1, CORIG, N1)
  const float* Wt    = (const float*)d_in[4];   // (OCH, CIN)
  const float* gamma = (const float*)d_in[5];   // (MID)
  const float* beta  = (const float*)d_in[6];   // (MID)
  const float* Ws    = (const float*)d_in[7];   // (OUT, NFCH)
  float* out = (float*)d_out;

  char* ws = (char*)d_ws;
  size_t off = 0;
  auto take = [&](size_t bytes) -> char* {
    char* p = ws + off;
    off = (off + bytes + 255) & ~(size_t)255;
    return p;
  };

  const size_t nWt = (size_t)OCH_ * CIN_;
  const size_t nF  = (size_t)B_ * L2_ * CIN_ * N2_;
  const size_t nWs = (size_t)OUT_ * NFCH_;
  const size_t nTf = (size_t)B_ * L2_ * OCH_ * N2_;
  const size_t nCh = (size_t)L2_ * OCH_;
  const size_t nAn = (size_t)B_ * L1_ * N1_;
  const size_t nNf = (size_t)B_ * L1_ * NFCH_ * N1_;

  unsigned short* Wt_h = (unsigned short*)take(nWt * 2);
  unsigned short* Ft_h = (unsigned short*)take(nF * 2);     // (N2 x CIN) per (b,l2)
  unsigned short* Ws_h = (unsigned short*)take(nWs * 2);
  float* tf            = (float*)take(nTf * 4);
  float* meanp         = (float*)take(nCh * 4);
  float* rstdp         = (float*)take(nCh * 4);
  int*   idxp          = (int*)take(nAn * 3 * 4);
  float* wgtp          = (float*)take(nAn * 3 * 4);
  unsigned short* nfT  = (unsigned short*)take(nNf * 2);    // (N1 x NFCH) per (b,t1)

  // 1) weights -> bf16; features -> bf16 transposed (B operand is N x K for TDM)
  f32_to_bf16_kernel<<<(unsigned)((nWt + 255) / 256), 256, 0, stream>>>(Wt, Wt_h, nWt);
  f32_to_bf16_kernel<<<(unsigned)((nWs + 255) / 256), 256, 0, stream>>>(Ws, Ws_h, nWs);
  transpose_to_bf16_kernel<<<dim3((CIN_ * N2_ + 255) / 256, 1, B_ * L2_), 256, 0, stream>>>(
      feats, Ft_h, CIN_, N2_);

  // 2) temporal conv GEMM: per (b,l2), tf = Wt(768x512) x F(512x512)
  wmma_gemm_bf16_kernel<<<dim3(N2_ / 128, OCH_ / 64, B_ * L2_), 256, 0, stream>>>(
      Wt_h, Ft_h, tf, OCH_, N2_, CIN_,
      0LL, (long long)CIN_ * N2_, (long long)OCH_ * N2_);

  // 3) BN batch stats per (t2,k,c) channel
  bn_stats_kernel<<<(unsigned)nCh, 256, 0, stream>>>(tf, meanp, rstdp);

  // 4) 3-NN indices + inverse-distance weights
  three_nn_kernel<<<(unsigned)((nAn + 255) / 256), 256, 0, stream>>>(oxyzs, xyzs, idxp, wgtp);

  // 5) fused BN/ReLU + interpolation + concat -> nfT (bf16, transposed)
  build_nf_kernel<<<(unsigned)((nNf + 255) / 256), 256, 0, stream>>>(
      tf, meanp, rstdp, gamma, beta, ofeat, idxp, wgtp, nfT);

  // 6) first tuple output: original_xyzs passthrough
  hipMemcpyAsync(out, oxyzs, (size_t)B_ * L1_ * N1_ * 3 * sizeof(float),
                 hipMemcpyDeviceToDevice, stream);

  // 7) spatial conv GEMM: per (b,t1), out = Ws(512x384) x nf(384x2048)
  float* outNF = out + (size_t)B_ * L1_ * N1_ * 3;
  wmma_gemm_bf16_kernel<<<dim3(N1_ / 128, OUT_ / 64, B_ * L1_), 256, 0, stream>>>(
      Ws_h, nfT, outNF, OUT_, N1_, NFCH_,
      0LL, (long long)NFCH_ * N1_, (long long)OUT_ * N1_);
}
